// TransformerModule_37142877176422
// MI455X (gfx1250) — compile-verified, matrix operand order checked by reading, not run on hardware
//
#include <hip/hip_runtime.h>
#include <hip/hip_bf16.h>

// ---------------------------------------------------------------------------
// Types for CDNA5 WMMA (wave32): D(16x16,f32) = A(16x32,bf16) x B(32x16,bf16) + C
// ---------------------------------------------------------------------------
typedef __attribute__((ext_vector_type(8)))  unsigned short v8us;
typedef __attribute__((ext_vector_type(16))) unsigned short v16us;
typedef __attribute__((ext_vector_type(16))) __bf16        v16bf;
typedef __attribute__((ext_vector_type(8)))  float         v8f;

#define DIMC 256
#define HIDC 1024

__device__ __forceinline__ unsigned short f2bf(float f) {
  unsigned int u = __builtin_bit_cast(unsigned int, f);
  u += 0x7fffu + ((u >> 16) & 1u);   // round-to-nearest-even
  return (unsigned short)(u >> 16);
}

__device__ __forceinline__ v8f zero8() {
  v8f z = {0.f, 0.f, 0.f, 0.f, 0.f, 0.f, 0.f, 0.f};
  return z;
}

// Load one 16x32 bf16 fragment half-row: lane holds row (lane&15), K offsets
// koff..koff+7 and koff+16..koff+23 where koff = (lane>>4)*8. `p` must already
// point at row_base + koff.
__device__ __forceinline__ v16bf load_frag(const unsigned short* p) {
  v8us lo = *(const v8us*)(p);
  v8us hi = *(const v8us*)(p + 16);
  v16us c = __builtin_shufflevector(lo, hi, 0, 1, 2, 3, 4, 5, 6, 7,
                                            8, 9, 10, 11, 12, 13, 14, 15);
  return __builtin_bit_cast(v16bf, c);
}

__device__ __forceinline__ v8f wmma_bf16(v16bf a, v16bf b, v8f c) {
  return __builtin_amdgcn_wmma_f32_16x16x32_bf16(
      /*neg_a=*/false, a, /*neg_b=*/false, b,
      /*c_mod=*/(short)0, c, /*reuse_a=*/false, /*reuse_b=*/false);
}

// Window-partition permutation (with roll by -shift): output token index
// (window-major order over 4 images of 64x64, 8x8 windows) -> source linear
// token index b*4096 + h*64 + w.
__device__ __forceinline__ int winperm_src(int tOut, int shift) {
  int wi = tOut >> 6, t = tOut & 63;
  int b = wi >> 6, wrem = wi & 63;
  int wh = wrem >> 3, ww = wrem & 7;
  int th = t >> 3, tw = t & 7;
  int h = (wh * 8 + th + shift) & 63;
  int w = (ww * 8 + tw + shift) & 63;
  return b * 4096 + h * 64 + w;
}

// ---------------------------------------------------------------------------
// LayerNorm (C=256): one wave per token, lane handles 8 channels.
// permShift < 0: identity gather; else window-partition gather with roll.
// Output bf16, row-major [ntok][256].
// ---------------------------------------------------------------------------
__global__ void __launch_bounds__(256) ln_kernel(
    unsigned short* __restrict__ out, const float* __restrict__ in,
    const float* __restrict__ g, const float* __restrict__ be,
    int ntok, int permShift) {
  int lane = threadIdx.x & 31;
  int wv = (int)((blockIdx.x * blockDim.x + threadIdx.x) >> 5);
  if (wv >= ntok) return;
  int src = (permShift < 0) ? wv : winperm_src(wv, permShift);
  const float4* p = (const float4*)(in + (size_t)src * DIMC + lane * 8);
  float4 a0 = p[0], a1 = p[1];
  float x[8] = {a0.x, a0.y, a0.z, a0.w, a1.x, a1.y, a1.z, a1.w};
  float s = 0.f, s2 = 0.f;
#pragma unroll
  for (int i = 0; i < 8; i++) { s += x[i]; s2 += x[i] * x[i]; }
#pragma unroll
  for (int off = 16; off >= 1; off >>= 1) {
    s += __shfl_xor(s, off);
    s2 += __shfl_xor(s2, off);
  }
  float mu = s * (1.f / 256.f);
  float var = s2 * (1.f / 256.f) - mu * mu;
  float inv = rsqrtf(var + 1e-5f);
  int c0 = lane * 8;
  v8us o;
#pragma unroll
  for (int i = 0; i < 8; i++)
    o[i] = f2bf((x[i] - mu) * inv * g[c0 + i] + be[c0 + i]);
  *(v8us*)(out + (size_t)wv * DIMC + c0) = o;
}

// ---------------------------------------------------------------------------
// WMMA GEMM: Y[m][n] = sum_k X[m][k] * W[n][k] + bias[n]   (X: bf16 row-major,
// W: bf16 row-major (out,in)).  One wave computes a 32x64 tile. The K-loop is
// unrolled 2x with ping-pong fragment buffers (compute on A while B's loads
// are in flight and vice versa) -> no register rotation copies.
// mode 0: bf16 row-major, scaled by alpha
// mode 1: bf16 transposed (out[n * tStride + m])  -- for V^T
// mode 2: exact GELU then bf16 row-major
// mode 3: fp32 residual add (out_f32[perm(m) * Nout + n] += v)
// ---------------------------------------------------------------------------
__global__ void __launch_bounds__(256) gemm_kernel(
    const unsigned short* __restrict__ X, const unsigned short* __restrict__ Wm,
    const float* __restrict__ bias, int Ntok, int Nout, int K, float alpha,
    int mode, int permShift, unsigned short* __restrict__ out_us,
    float* __restrict__ out_f32, int tStride) {
  int lane = threadIdx.x & 31;
  int wid = (int)((blockIdx.x * blockDim.x + threadIdx.x) >> 5);
  int nGroups = Nout >> 6;
  int mt = wid / nGroups, ng = wid - mt * nGroups;
  int m0 = mt << 5, n0 = ng << 6;
  if (m0 >= Ntok) return;
  int nlow = lane & 15, koff = (lane >> 4) << 3, half = lane >> 4;
  const unsigned short* xrow0 = X + (size_t)(m0 + nlow) * K + koff;
  const unsigned short* xrow1 = X + (size_t)(m0 + 16 + nlow) * K + koff;
  const unsigned short* wr0 = Wm + (size_t)(n0 + nlow) * K + koff;
  v8f acc[2][4];
#pragma unroll
  for (int mi = 0; mi < 2; mi++)
#pragma unroll
    for (int j = 0; j < 4; j++) acc[mi][j] = zero8();

  v16bf aa[2][2];   // [buffer][m-subtile]
  v16bf bb[2][4];   // [buffer][n-subtile]

  auto loadset = [&](int buf, int kk) {
    aa[buf][0] = load_frag(xrow0 + kk);
    aa[buf][1] = load_frag(xrow1 + kk);
#pragma unroll
    for (int j = 0; j < 4; j++)
      bb[buf][j] = load_frag(wr0 + (size_t)(j * 16) * K + kk);
  };
  auto compute = [&](int buf) {
#pragma unroll
    for (int j = 0; j < 4; j++)
      acc[0][j] = wmma_bf16(aa[buf][0], bb[buf][j], acc[0][j]);
#pragma unroll
    for (int j = 0; j < 4; j++)
      acc[1][j] = wmma_bf16(aa[buf][1], bb[buf][j], acc[1][j]);
  };

  // K is a multiple of 64 (256 or 1024): exact 2x32 ping-pong with peeled tail
  loadset(0, 0);
  int k0 = 0;
  for (; k0 + 64 < K; k0 += 64) {
    loadset(1, k0 + 32);
    __builtin_prefetch(xrow0 + k0 + 128, 0, 1);  // global_prefetch_b8
    compute(0);
    loadset(0, k0 + 64);
    compute(1);
  }
  loadset(1, k0 + 32);
  compute(0);
  compute(1);

#pragma unroll
  for (int mi = 0; mi < 2; mi++) {
#pragma unroll
    for (int j = 0; j < 4; j++) {
      int n = n0 + j * 16 + nlow;
      float bv = bias[n];
#pragma unroll
      for (int r = 0; r < 8; r++) {
        int m = m0 + mi * 16 + r + half * 8;
        float v = acc[mi][j][r] + bv;
        if (mode == 0) {
          out_us[(size_t)m * Nout + n] = f2bf(v * alpha);
        } else if (mode == 1) {
          out_us[(size_t)n * tStride + m] = f2bf(v);
        } else if (mode == 2) {
          float gv = 0.5f * v * (1.0f + erff(v * 0.70710678118654752f));
          out_us[(size_t)m * Nout + n] = f2bf(gv);
        } else {
          int lin = (permShift < 0) ? m : winperm_src(m, permShift);
          out_f32[(size_t)lin * Nout + n] += v;
        }
      }
    }
  }
}

// ---------------------------------------------------------------------------
// Windowed self-attention (T=64, C=256). One wave = 16 queries of one window.
// Q pre-scaled by C^-0.5 (folded into Q GEMM). P staged through LDS as bf16.
// ---------------------------------------------------------------------------
__global__ void __launch_bounds__(256) winattn_kernel(
    const unsigned short* __restrict__ Q, const unsigned short* __restrict__ Kt,
    const unsigned short* __restrict__ Vt, unsigned short* __restrict__ O,
    int vtStride, int useMask) {
  __shared__ unsigned short plds[8 * 16 * 64];
  int lane = threadIdx.x & 31, wv = threadIdx.x >> 5;
  int wid = blockIdx.x * 8 + wv;
  int win = wid >> 2, qt = wid & 3;
  int q0 = win * 64 + qt * 16;
  int nlow = lane & 15, koff = (lane >> 4) << 3, half = lane >> 4;
  const unsigned short* qrow = Q + (size_t)(q0 + nlow) * DIMC + koff;
  v16bf qa[8];
#pragma unroll
  for (int kk = 0; kk < 8; kk++) qa[kk] = load_frag(qrow + kk * 32);
  v8f s[4];
#pragma unroll
  for (int nt = 0; nt < 4; nt++) {
    v8f a = zero8();
    const unsigned short* krow =
        Kt + (size_t)(win * 64 + nt * 16 + nlow) * DIMC + koff;
#pragma unroll
    for (int kk = 0; kk < 8; kk++)
      a = wmma_bf16(qa[kk], load_frag(krow + kk * 32), a);
    s[nt] = a;
  }
  if (useMask) {  // shifted-window region mask (-100 when regions differ)
    int wrem = win & 63, wh = wrem >> 3, ww = wrem & 7;
#pragma unroll
    for (int nt = 0; nt < 4; nt++) {
      int tk = nt * 16 + nlow;
      int hk = wh * 8 + (tk >> 3), wk = ww * 8 + (tk & 7);
      int ck = (hk < 56 ? 0 : (hk < 60 ? 1 : 2)) * 3 +
               (wk < 56 ? 0 : (wk < 60 ? 1 : 2));
#pragma unroll
      for (int r = 0; r < 8; r++) {
        int tq = qt * 16 + r + half * 8;
        int hq = wh * 8 + (tq >> 3), wq = ww * 8 + (tq & 7);
        int cq = (hq < 56 ? 0 : (hq < 60 ? 1 : 2)) * 3 +
                 (wq < 56 ? 0 : (wq < 60 ? 1 : 2));
        if (cq != ck) s[nt][r] -= 100.0f;
      }
    }
  }
  int pbase = wv * 1024;
#pragma unroll
  for (int r = 0; r < 8; r++) {
    float mx = fmaxf(fmaxf(s[0][r], s[1][r]), fmaxf(s[2][r], s[3][r]));
    mx = fmaxf(mx, __shfl_xor(mx, 1));
    mx = fmaxf(mx, __shfl_xor(mx, 2));
    mx = fmaxf(mx, __shfl_xor(mx, 4));
    mx = fmaxf(mx, __shfl_xor(mx, 8));
    float rs = 0.f;
#pragma unroll
    for (int nt = 0; nt < 4; nt++) {
      float e = __expf(s[nt][r] - mx);
      s[nt][r] = e;
      rs += e;
    }
    rs += __shfl_xor(rs, 1);
    rs += __shfl_xor(rs, 2);
    rs += __shfl_xor(rs, 4);
    rs += __shfl_xor(rs, 8);
    float rinv = 1.0f / rs;
#pragma unroll
    for (int nt = 0; nt < 4; nt++)
      plds[pbase + (r + half * 8) * 64 + nt * 16 + nlow] = f2bf(s[nt][r] * rinv);
  }
  // O = P @ V  (B fragments from V^T so loads are contiguous)
#pragma unroll
  for (int ct = 0; ct < 16; ct++) {
    v8f o = zero8();
#pragma unroll
    for (int ch = 0; ch < 2; ch++) {
      v8us lo = *(const v8us*)&plds[pbase + nlow * 64 + ch * 32 + koff];
      v8us hi = *(const v8us*)&plds[pbase + nlow * 64 + ch * 32 + koff + 16];
      v16us cc = __builtin_shufflevector(lo, hi, 0, 1, 2, 3, 4, 5, 6, 7,
                                                 8, 9, 10, 11, 12, 13, 14, 15);
      v16bf pa = __builtin_bit_cast(v16bf, cc);
      const unsigned short* vr =
          Vt + (size_t)(ct * 16 + nlow) * vtStride + win * 64 + ch * 32 + koff;
      o = wmma_bf16(pa, load_frag(vr), o);
    }
#pragma unroll
    for (int r = 0; r < 8; r++)
      O[(size_t)(q0 + r + half * 8) * DIMC + ct * 16 + nlow] = f2bf(o[r]);
  }
}

// ---------------------------------------------------------------------------
// Cross attention, flash style (T=4096 keys per batch, B=2). One wave = 16
// queries; streams 32-key chunks with online softmax; P staged via LDS.
// Token layout: [batch*4096 + t][256]; V is transposed [256][8192].
// ---------------------------------------------------------------------------
__global__ void __launch_bounds__(256) crossattn_kernel(
    const unsigned short* __restrict__ Q, const unsigned short* __restrict__ Kt,
    const unsigned short* __restrict__ Vt, unsigned short* __restrict__ O) {
  __shared__ unsigned short plds[8 * 16 * 32];
  int lane = threadIdx.x & 31, wv = threadIdx.x >> 5;
  int wid = blockIdx.x * 8 + wv;
  int batch = wid >> 8, qt = wid & 255;
  int q0 = batch * 4096 + qt * 16;
  int nlow = lane & 15, koff = (lane >> 4) << 3, half = lane >> 4;
  const unsigned short* qrow = Q + (size_t)(q0 + nlow) * DIMC + koff;
  v16bf qa[8];
#pragma unroll
  for (int kk = 0; kk < 8; kk++) qa[kk] = load_frag(qrow + kk * 32);
  v8f out[16];
#pragma unroll
  for (int t = 0; t < 16; t++) out[t] = zero8();
  float mrun[8], lrun[8];
#pragma unroll
  for (int r = 0; r < 8; r++) { mrun[r] = -1e30f; lrun[r] = 0.f; }
  int pbase = wv * 512;
  for (int kc = 0; kc < 128; kc++) {
    int kb0 = batch * 4096 + kc * 32;
    v8f s0 = zero8(), s1 = zero8();
    const unsigned short* k0p = Kt + (size_t)(kb0 + nlow) * DIMC + koff;
    const unsigned short* k1p = Kt + (size_t)(kb0 + 16 + nlow) * DIMC + koff;
    __builtin_prefetch(k0p + 32 * DIMC, 0, 1);  // next key chunk
#pragma unroll
    for (int kk = 0; kk < 8; kk++) {
      s0 = wmma_bf16(qa[kk], load_frag(k0p + kk * 32), s0);
      s1 = wmma_bf16(qa[kk], load_frag(k1p + kk * 32), s1);
    }
    float sc[8];
#pragma unroll
    for (int r = 0; r < 8; r++) {
      float cm = fmaxf(s0[r], s1[r]);
      cm = fmaxf(cm, __shfl_xor(cm, 1));
      cm = fmaxf(cm, __shfl_xor(cm, 2));
      cm = fmaxf(cm, __shfl_xor(cm, 4));
      cm = fmaxf(cm, __shfl_xor(cm, 8));
      float nm = fmaxf(mrun[r], cm);
      sc[r] = __expf(mrun[r] - nm);
      mrun[r] = nm;
      float e0 = __expf(s0[r] - nm), e1 = __expf(s1[r] - nm);
      float rs = e0 + e1;
      rs += __shfl_xor(rs, 1);
      rs += __shfl_xor(rs, 2);
      rs += __shfl_xor(rs, 4);
      rs += __shfl_xor(rs, 8);
      lrun[r] = lrun[r] * sc[r] + rs;
      plds[pbase + (r + half * 8) * 32 + nlow] = f2bf(e0);
      plds[pbase + (r + half * 8) * 32 + 16 + nlow] = f2bf(e1);
    }
#pragma unroll
    for (int t = 0; t < 16; t++)
#pragma unroll
      for (int r = 0; r < 8; r++) out[t][r] *= sc[r];
    v8us lo = *(const v8us*)&plds[pbase + nlow * 32 + koff];
    v8us hi = *(const v8us*)&plds[pbase + nlow * 32 + koff + 16];
    v16us cc = __builtin_shufflevector(lo, hi, 0, 1, 2, 3, 4, 5, 6, 7,
                                               8, 9, 10, 11, 12, 13, 14, 15);
    v16bf pa = __builtin_bit_cast(v16bf, cc);
#pragma unroll
    for (int ct = 0; ct < 16; ct++) {
      const unsigned short* vr =
          Vt + (size_t)(ct * 16 + nlow) * 8192 + kb0 + koff;
      out[ct] = wmma_bf16(pa, load_frag(vr), out[ct]);
    }
  }
  float rinv[8];
#pragma unroll
  for (int r = 0; r < 8; r++) rinv[r] = 1.0f / lrun[r];
#pragma unroll
  for (int ct = 0; ct < 16; ct++)
#pragma unroll
    for (int r = 0; r < 8; r++)
      O[(size_t)(q0 + r + half * 8) * DIMC + ct * 16 + nlow] =
          f2bf(out[ct][r] * rinv[r]);
}

__global__ void __launch_bounds__(256) add_kernel(float* __restrict__ dst,
                                                  const float* __restrict__ a,
                                                  const float* __restrict__ b,
                                                  int n) {
  int i = blockIdx.x * blockDim.x + threadIdx.x;
  if (i < n) dst[i] = a[i] + b[i];
}

__global__ void __launch_bounds__(256) cvt_kernel(unsigned short* __restrict__ d,
                                                  const float* __restrict__ s,
                                                  int n) {
  int i = blockIdx.x * blockDim.x + threadIdx.x;
  if (i < n) d[i] = f2bf(s[i]);
}

// ---------------------------------------------------------------------------
// Host orchestration
// ---------------------------------------------------------------------------
struct BP {
  const float *bn1, *bn2, *bn3, *g1, *g2, *g3;
  const float *kb, *kw, *m1b, *m1w, *m2b, *m2w, *pb, *pw, *qb, *qw, *vb, *vw;
};

static BP load_bp(void* const* din, int base, bool ln3) {
  BP p;
  int i = base;
  p.bn1 = (const float*)din[i++];
  p.bn2 = (const float*)din[i++];
  p.bn3 = ln3 ? (const float*)din[i++] : nullptr;
  p.g1 = (const float*)din[i++];
  p.g2 = (const float*)din[i++];
  p.g3 = ln3 ? (const float*)din[i++] : nullptr;
  p.kb = (const float*)din[i++];
  p.kw = (const float*)din[i++];
  p.m1b = (const float*)din[i++];
  p.m1w = (const float*)din[i++];
  p.m2b = (const float*)din[i++];
  p.m2w = (const float*)din[i++];
  p.pb = (const float*)din[i++];
  p.pw = (const float*)din[i++];
  p.qb = (const float*)din[i++];
  p.qw = (const float*)din[i++];
  p.vb = (const float*)din[i++];
  p.vw = (const float*)din[i++];
  return p;
}

struct Ctx {
  float* fmap;          // [16384][256] fp32 (layer state, concat of fmap1|fmap2)
  float* xbuf;          // [8192][256] fp32 cross-block scratch
  unsigned short* xln;  // [16384][256] bf16
  unsigned short* q;    // [16384][256] bf16
  unsigned short* k;    // [16384][256] bf16
  unsigned short* vt;   // [256][16384] bf16 (transposed V)
  unsigned short* att;  // [16384][256] bf16 (attn out / 2nd LN buffer)
  unsigned short* hid;  // [16384][1024] bf16 + weight staging tail
  hipStream_t stream;
};

static void run_gemm(Ctx& c, const unsigned short* X, const unsigned short* W,
                     const float* b, int Ntok, int Nout, int K, float alpha,
                     int mode, int permShift, unsigned short* ous, float* of32,
                     int tStride) {
  int waves = (Ntok / 32) * (Nout / 64);
  int blocks = (waves + 7) / 8;
  gemm_kernel<<<blocks, 256, 0, c.stream>>>(X, W, b, Ntok, Nout, K, alpha,
                                            mode, permShift, ous, of32,
                                            tStride);
}

static unsigned short* stage_w(Ctx& c, const float* w, int n, int slot) {
  // staging area: after hid's 16384*1024 region, 3 slots of 256K bf16 elems
  unsigned short* base = c.hid + (size_t)16384 * 1024 + (size_t)slot * 262144;
  cvt_kernel<<<(n + 255) / 256, 256, 0, c.stream>>>(base, w, n);
  return base;
}

static void self_block(Ctx& c, const BP& p, int shift, int useMask) {
  const int N = 16384;
  ln_kernel<<<N / 8, 256, 0, c.stream>>>(c.xln, c.fmap, p.g1, p.bn1, N, shift);
  unsigned short* wq = stage_w(c, p.qw, 65536, 0);
  unsigned short* wk = stage_w(c, p.kw, 65536, 1);
  unsigned short* wv = stage_w(c, p.vw, 65536, 2);
  run_gemm(c, c.xln, wq, p.qb, N, 256, 256, 0.0625f, 0, -1, c.q, nullptr, 0);
  run_gemm(c, c.xln, wk, p.kb, N, 256, 256, 1.0f, 0, -1, c.k, nullptr, 0);
  run_gemm(c, c.xln, wv, p.vb, N, 256, 256, 1.0f, 1, -1, c.vt, nullptr, 16384);
  winattn_kernel<<<128, 256, 0, c.stream>>>(c.q, c.k, c.vt, c.att, 16384,
                                            useMask);
  unsigned short* wp = stage_w(c, p.pw, 65536, 0);
  run_gemm(c, c.att, wp, p.pb, N, 256, 256, 1.0f, 3, shift, nullptr, c.fmap, 0);
  ln_kernel<<<N / 8, 256, 0, c.stream>>>(c.xln, c.fmap, p.g2, p.bn2, N, -1);
  unsigned short* w1 = stage_w(c, p.m1w, 262144, 0);
  run_gemm(c, c.xln, w1, p.m1b, N, 1024, 256, 1.0f, 2, -1, c.hid, nullptr, 0);
  unsigned short* w2 = stage_w(c, p.m2w, 262144, 1);
  run_gemm(c, c.hid, w2, p.m2b, N, 256, 1024, 1.0f, 3, -1, nullptr, c.fmap, 0);
}

// cross_block(shortcut, kv): Q from LN(shortcut,g1), K/V from LN(kv,g2);
// xbuf accumulates shortcut + proj(attn) + mlp; final shortcut = xbuf + shortcut.
static void cross_pass(Ctx& c, float* shortcut, const float* kv, const BP& p) {
  const int N = 8192;
  hipMemcpyAsync(c.xbuf, shortcut, (size_t)N * 256 * sizeof(float),
                 hipMemcpyDeviceToDevice, c.stream);
  ln_kernel<<<N / 8, 256, 0, c.stream>>>(c.xln, shortcut, p.g1, p.bn1, N, -1);
  ln_kernel<<<N / 8, 256, 0, c.stream>>>(c.att, kv, p.g2, p.bn2, N, -1);
  unsigned short* wq = stage_w(c, p.qw, 65536, 0);
  unsigned short* wk = stage_w(c, p.kw, 65536, 1);
  unsigned short* wv = stage_w(c, p.vw, 65536, 2);
  run_gemm(c, c.xln, wq, p.qb, N, 256, 256, 0.0625f, 0, -1, c.q, nullptr, 0);
  run_gemm(c, c.att, wk, p.kb, N, 256, 256, 1.0f, 0, -1, c.k, nullptr, 0);
  run_gemm(c, c.att, wv, p.vb, N, 256, 256, 1.0f, 1, -1, c.vt, nullptr, 8192);
  crossattn_kernel<<<64, 256, 0, c.stream>>>(c.q, c.k, c.vt, c.att);
  unsigned short* wp = stage_w(c, p.pw, 65536, 0);
  run_gemm(c, c.att, wp, p.pb, N, 256, 256, 1.0f, 3, -1, nullptr, c.xbuf, 0);
  ln_kernel<<<N / 8, 256, 0, c.stream>>>(c.xln, c.xbuf, p.g3, p.bn3, N, -1);
  unsigned short* w1 = stage_w(c, p.m1w, 262144, 0);
  run_gemm(c, c.xln, w1, p.m1b, N, 1024, 256, 1.0f, 2, -1, c.hid, nullptr, 0);
  unsigned short* w2 = stage_w(c, p.m2w, 262144, 1);
  run_gemm(c, c.hid, w2, p.m2b, N, 256, 1024, 1.0f, 3, -1, nullptr, c.xbuf, 0);
  int n = N * 256;
  add_kernel<<<(n + 255) / 256, 256, 0, c.stream>>>(shortcut, c.xbuf, shortcut,
                                                    n);
}

extern "C" void kernel_launch(void* const* d_in, const int* in_sizes, int n_in,
                              void* d_out, int out_size, void* d_ws,
                              size_t ws_size, hipStream_t stream) {
  (void)in_sizes;
  (void)n_in;
  (void)out_size;
  (void)ws_size;
  char* ws = (char*)d_ws;
  Ctx c;
  c.stream = stream;
  size_t o = 0;
  c.fmap = (float*)(ws + o);          o += (size_t)16384 * 256 * 4;
  c.xbuf = (float*)(ws + o);          o += (size_t)8192 * 256 * 4;
  c.xln = (unsigned short*)(ws + o);  o += (size_t)16384 * 256 * 2;
  c.q = (unsigned short*)(ws + o);    o += (size_t)16384 * 256 * 2;
  c.k = (unsigned short*)(ws + o);    o += (size_t)16384 * 256 * 2;
  c.vt = (unsigned short*)(ws + o);   o += (size_t)16384 * 256 * 2;
  c.att = (unsigned short*)(ws + o);  o += (size_t)16384 * 256 * 2;
  c.hid = (unsigned short*)(ws + o);  // 16384*1024*2 + 3*262144*2 staging

  // params: top-level sorted keys {fmap1, fmap2, params}; jax pytrees flatten
  // dicts in sorted-key order, lists in order.
  int idx = 2;
  BP l0b0 = load_bp(d_in, idx, false); idx += 16;
  BP l0b1 = load_bp(d_in, idx, false); idx += 16;
  BP l0c  = load_bp(d_in, idx, true);  idx += 18;
  BP l1b0 = load_bp(d_in, idx, false); idx += 16;
  BP l1b1 = load_bp(d_in, idx, false); idx += 16;

  // fmap = concat(fmap1, fmap2)
  hipMemcpyAsync(c.fmap, d_in[0], (size_t)8192 * 256 * 4,
                 hipMemcpyDeviceToDevice, stream);
  hipMemcpyAsync(c.fmap + (size_t)8192 * 256, d_in[1], (size_t)8192 * 256 * 4,
                 hipMemcpyDeviceToDevice, stream);

  // layer 0: two self blocks (non-shifted, shifted) then cross
  self_block(c, l0b0, /*shift=*/0, /*mask=*/0);
  self_block(c, l0b1, /*shift=*/4, /*mask=*/1);
  float* f1 = c.fmap;
  float* f2 = c.fmap + (size_t)8192 * 256;
  cross_pass(c, f2, f1, l0c);  // fmap2 = cross_block(f2, f1) + f2
  cross_pass(c, f1, f2, l0c);  // fmap1 = cross_block(f1, f2') + f1

  // layer 1: two self blocks, no cross
  self_block(c, l1b0, 0, 0);
  self_block(c, l1b1, 4, 1);

  hipMemcpyAsync(d_out, c.fmap, (size_t)16384 * 256 * 4,
                 hipMemcpyDeviceToDevice, stream);
}
